// SuperTitansViT_33818572488842
// MI455X (gfx1250) — compile-verified, matrix operand order checked by reading, not run on hardware
//
#include <hip/hip_runtime.h>
#include <hip/hip_bf16.h>
#include <math.h>
#include <stdint.h>

typedef __bf16 bf16_t;
typedef __attribute__((ext_vector_type(16))) __bf16 v16bf;
typedef __attribute__((ext_vector_type(8)))  float  v8f;

// ---------------------------------------------------------------- constants
#define Bsz   32
#define Dm    768
#define Hh    12
#define DH    64
#define Im    768
#define Ltot  12
#define NTOK  197
#define NWIN  15
#define WINW  14
#define NATT  210      // 15*14
#define NMEM  256      // padded memory tokens
#define NVAL  201      // 197 + 4 registers
#define CHm   64
#define NCH   4

// ---------------------------------------------------------------- helpers
__device__ inline float gelu_f(float x) {
  const float k0 = 0.7978845608028654f, k1 = 0.044715f;
  float t = tanhf(k0 * (x + k1 * x * x * x));
  return 0.5f * x * (1.f + t);
}
__device__ inline float gelu_grad_f(float x) {
  const float k0 = 0.7978845608028654f, k1 = 0.044715f;
  float u = k0 * (x + k1 * x * x * x);
  float t = tanhf(u);
  float du = k0 * (1.f + 3.f * k1 * x * x);
  return 0.5f * (1.f + t) + 0.5f * x * (1.f - t * t) * du;
}
__device__ inline float sigmoid_f(float x) { return 1.f / (1.f + __expf(-x)); }

// block reduction over 256 threads (8 wave32s)
__device__ inline float block_sum(float v, float* red) {
  for (int o = 16; o; o >>= 1) v += __shfl_xor(v, o, 32);
  int lane = threadIdx.x & 31, wave = threadIdx.x >> 5;
  if (lane == 0) red[wave] = v;
  __syncthreads();
  if (threadIdx.x == 0) {
    float s = red[0];
    for (int i = 1; i < 8; ++i) s += red[i];
    red[0] = s;
  }
  __syncthreads();
  float r = red[0];
  __syncthreads();
  return r;
}

// ----------------------------------------------- CDNA5 async global->LDS
__device__ inline void async_copy_b128(void* lds, const void* g) {
  asm volatile("global_load_async_to_lds_b128 %0, %1, off"
               :: "v"((uint32_t)(uintptr_t)lds), "v"(g) : "memory");
}
__device__ inline void wait_asynccnt0() {
  asm volatile("s_wait_asynccnt 0x0" ::: "memory");
}

// --------------------------------------------------------- WMMA fragments
// contiguous 16x32 bf16 fragment: "row" (M for A, N for transposed-B tile)
// = lane&15, lane>>4 selects k 0-7,16-23 (h=0) or 8-15,24-31 (h=1), per the CDNA5 16-bit A layout.
__device__ inline v16bf frag_row(const bf16_t* base, int ld, int lane) {
  const bf16_t* p = base + (lane & 15) * ld + ((lane >> 4) << 3);
  v16bf f;
#pragma unroll
  for (int e = 0; e < 8; ++e) f[e] = p[e];
#pragma unroll
  for (int e = 0; e < 8; ++e) f[8 + e] = p[16 + e];
  return f;
}
// transpose-capable fragment loaders for the 64x64 block matmul
__device__ inline v16bf frag_a(const bf16_t* A, int lda, bool tA, int m0, int k0, int lane) {
  int m = m0 + (lane & 15);
  int kh = k0 + ((lane >> 4) << 3);
  v16bf f;
#pragma unroll
  for (int e = 0; e < 16; ++e) {
    int k = kh + (e < 8 ? e : e + 8);
    f[e] = tA ? A[k * lda + m] : A[m * lda + k];
  }
  return f;
}
__device__ inline v16bf frag_b(const bf16_t* B, int ldb, bool tB, int k0, int n0, int lane) {
  int n = n0 + (lane & 15);
  int kh = k0 + ((lane >> 4) << 3);
  v16bf f;
#pragma unroll
  for (int e = 0; e < 16; ++e) {
    int k = kh + (e < 8 ? e : e + 8);
    f[e] = tB ? B[n * ldb + k] : B[k * ldb + n];
  }
  return f;
}
__device__ inline v8f wmma_bf16(v16bf a, v16bf b, v8f c) {
  return __builtin_amdgcn_wmma_f32_16x16x32_bf16(false, a, false, b, (short)0, c, false, false);
}

// ---------------------------------------------------------------- GEMM
// C[M,N] = A[M,K] @ B[K,N] (+bias[N]); f32 in/out, bf16 WMMA core.
// 256 threads = 8 waves (4x2); block tile 128x64; each wave does a 32x32
// macro-tile = 4 WMMAs per K-step with 2x fragment reuse. B tile staged
// transposed so both fragment reads are contiguous b128 LDS loads.
template <bool ADD_BIAS>
__global__ void gemm_wmma(const float* __restrict__ A, const float* __restrict__ B,
                          const float* __restrict__ bias, float* __restrict__ C,
                          int M, int N, int K) {
  constexpr int MT = 128, NT = 64, KT = 32, LDT = KT + 8;
  __shared__ bf16_t As[MT * LDT];
  __shared__ bf16_t Bt[NT * LDT];
  int wave = threadIdx.x >> 5, lane = threadIdx.x & 31;
  int wm = wave >> 1, wn = wave & 1;
  int m0 = blockIdx.y * MT, n0 = blockIdx.x * NT;
  v8f acc00 = {}, acc01 = {}, acc10 = {}, acc11 = {};
  for (int k0 = 0; k0 < K; k0 += KT) {
    for (int i = threadIdx.x; i < MT * KT; i += 256) {
      int r = i >> 5, c = i & 31;
      int gm = m0 + r, gk = k0 + c;
      As[r * LDT + c] = (bf16_t)((gm < M && gk < K) ? A[(size_t)gm * K + gk] : 0.f);
    }
    for (int i = threadIdx.x; i < NT * KT; i += 256) {
      int n = i & 63, c = i >> 6;
      int gn = n0 + n, gk = k0 + c;
      Bt[n * LDT + c] = (bf16_t)((gn < N && gk < K) ? B[(size_t)gk * N + gn] : 0.f);
    }
    if (k0 + KT < K) {
      int gm = m0 + (int)threadIdx.x;
      if (gm < M && (int)threadIdx.x < MT)
        __builtin_prefetch(&A[(size_t)gm * K + k0 + KT], 0, 1);
    }
    __syncthreads();
    v16bf fa0 = frag_row(As + (wm * 32 + 0) * LDT, LDT, lane);
    v16bf fa1 = frag_row(As + (wm * 32 + 16) * LDT, LDT, lane);
    v16bf fb0 = frag_row(Bt + (wn * 32 + 0) * LDT, LDT, lane);
    v16bf fb1 = frag_row(Bt + (wn * 32 + 16) * LDT, LDT, lane);
    acc00 = wmma_bf16(fa0, fb0, acc00);
    acc01 = wmma_bf16(fa0, fb1, acc01);
    acc10 = wmma_bf16(fa1, fb0, acc10);
    acc11 = wmma_bf16(fa1, fb1, acc11);
    __syncthreads();
  }
  int nA = lane & 15;
  int mrow = (lane >> 4) << 3;
  auto store_tile = [&](const v8f& acc, int mt, int nt) {
    int n = n0 + wn * 32 + nt * 16 + nA;
    if (n >= N) return;
    int mb = m0 + wm * 32 + mt * 16 + mrow;
#pragma unroll
    for (int r = 0; r < 8; ++r) {
      int m = mb + r;
      if (m < M) {
        float v = acc[r];
        if (ADD_BIAS) v += bias[n];
        C[(size_t)m * N + n] = v;
      }
    }
  };
  store_tile(acc00, 0, 0);
  store_tile(acc01, 0, 1);
  store_tile(acc10, 1, 0);
  store_tile(acc11, 1, 1);
}

// ------------------------------------------------------------ layernorm
// grid.x = row within batch (rows_out), grid.y = batch. Rows >= valid -> 0.
__global__ void layernorm_kernel(const float* __restrict__ x, const float* __restrict__ s,
                                 const float* __restrict__ bb, float* __restrict__ y,
                                 int rows_in, int rows_out, int valid, int d) {
  int r = blockIdx.x, b = blockIdx.y;
  float* orow = y + ((size_t)(b * rows_out + r)) * d;
  if (r >= valid) {
    for (int i = threadIdx.x; i < d; i += 256) orow[i] = 0.f;
    return;
  }
  const float* irow = x + ((size_t)(b * rows_in + r)) * d;
  __shared__ float buf[Dm];
  __shared__ float red[8];
  float ps = 0.f, pq = 0.f;
  for (int i = threadIdx.x; i < d; i += 256) {
    float v = irow[i];
    buf[i] = v;
    ps += v;
    pq += v * v;
  }
  float sum = block_sum(ps, red);
  float sq = block_sum(pq, red);
  float m = sum / d;
  float var = sq / d - m * m;
  float rs = rsqrtf(var + 1e-5f);
  for (int i = threadIdx.x; i < d; i += 256)
    orow[i] = (buf[i] - m) * rs * s[i] + bb[i];
}

// ------------------------------------------------------- patchify + LN1
__global__ void patchify_ln_kernel(const float* __restrict__ img, const float* __restrict__ s,
                                   const float* __restrict__ bb, float* __restrict__ out) {
  int p = blockIdx.x, b = blockIdx.y;
  int gi = p / 14, gj = p % 14;
  __shared__ float buf[Dm];
  __shared__ float red[8];
  float ps = 0.f, pq = 0.f;
  for (int i = threadIdx.x; i < Dm; i += 256) {
    int c = i % 3, pj = (i / 3) % 16, pi = i / 48;
    float v = img[((size_t)(b * 3 + c) * 224 + gi * 16 + pi) * 224 + gj * 16 + pj];
    buf[i] = v;
    ps += v;
    pq += v * v;
  }
  float sum = block_sum(ps, red);
  float sq = block_sum(pq, red);
  float m = sum / (float)Dm;
  float var = sq / (float)Dm - m * m;
  float rs = rsqrtf(var + 1e-5f);
  float* orow = out + ((size_t)(b * 196 + p)) * Dm;
  for (int i = threadIdx.x; i < Dm; i += 256)
    orow[i] = (buf[i] - m) * rs * s[i] + bb[i];
}

// --------------------------------------- LN2 + sincos posemb -> x tokens
__global__ void postpatch_kernel(const float* __restrict__ pe, const float* __restrict__ s,
                                 const float* __restrict__ bb, float* __restrict__ x) {
  int p = blockIdx.x, b = blockIdx.y;
  const float* irow = pe + ((size_t)(b * 196 + p)) * Dm;
  float* orow = x + ((size_t)(b * NTOK + 1 + p)) * Dm;
  __shared__ float buf[Dm];
  __shared__ float red[8];
  float ps = 0.f, pq = 0.f;
  for (int i = threadIdx.x; i < Dm; i += 256) {
    float v = irow[i];
    buf[i] = v;
    ps += v;
    pq += v * v;
  }
  float sum = block_sum(ps, red);
  float sq = block_sum(pq, red);
  float m = sum / (float)Dm;
  float var = sq / (float)Dm - m * m;
  float rs = rsqrtf(var + 1e-5f);
  int gy = p / 14, gx = p % 14;
  for (int i = threadIdx.x; i < Dm; i += 256) {
    int q = i / 192, ii = i % 192;
    float omega = __expf(-(float)ii * (9.210340371976184f / 191.f));
    float arg = (q < 2 ? (float)gx : (float)gy) * omega;
    float pev = ((q & 1) == 0) ? __sinf(arg) : __cosf(arg);
    orow[i] = (buf[i] - m) * rs * s[i] + bb[i] + pev;
  }
}

__global__ void write_cls_kernel(const float* __restrict__ cls, float* __restrict__ x) {
  int b = blockIdx.x;
  for (int i = threadIdx.x; i < Dm; i += 256)
    x[((size_t)b * NTOK) * Dm + i] = cls[i];
}

// ----------------------------------------------------- window attention
// q/k/v tiles staged with CDNA5 async global->LDS B128 copies (ASYNCcnt).
__global__ void window_attn_kernel(const float* __restrict__ qkv, float* __restrict__ out) {
  int w = blockIdx.x, h = blockIdx.y, b = blockIdx.z;
  __shared__ float qs[WINW * DH], ks[WINW * DH], vs[WINW * DH];
  __shared__ float dots[WINW * WINW];
  int row0 = b * NATT + w * WINW;
  int i = threadIdx.x;
  if (i < 224) {                       // 14 rows * 16 float4 segments
    int t = i >> 4;                    // row 0..13
    int c4 = (i & 15) << 2;            // col 0,4,...,60
    const float* g = qkv + (size_t)(row0 + t) * (3 * Im) + h * DH + c4;
    async_copy_b128(&qs[t * DH + c4], g);
    async_copy_b128(&ks[t * DH + c4], g + Im);
    async_copy_b128(&vs[t * DH + c4], g + 2 * Im);
  }
  wait_asynccnt0();
  __syncthreads();
  for (int j = threadIdx.x; j < WINW * WINW; j += 256) {
    int wq = j / WINW, wk = j % WINW;
    float s = 0.f;
    for (int d = 0; d < DH; ++d) s += qs[wq * DH + d] * ks[wk * DH + d];
    dots[j] = s * 0.125f;  // 64^-0.5
  }
  __syncthreads();
  if (threadIdx.x < WINW) {
    int r = threadIdx.x;
    float mx = dots[r * WINW];
    for (int j = 1; j < WINW; ++j) mx = fmaxf(mx, dots[r * WINW + j]);
    float se = 0.f;
    for (int j = 0; j < WINW; ++j) {
      float e = __expf(dots[r * WINW + j] - mx);
      dots[r * WINW + j] = e;
      se += e;
    }
    float inv = 1.f / se;
    for (int j = 0; j < WINW; ++j) dots[r * WINW + j] *= inv;
  }
  __syncthreads();
  for (int j = threadIdx.x; j < WINW * DH; j += 256) {
    int wq = j >> 6, d = j & 63;
    float s = 0.f;
    for (int wk = 0; wk < WINW; ++wk) s += dots[wq * WINW + wk] * vs[wk * DH + d];
    out[(size_t)(row0 + wq) * Im + h * DH + d] = s;
  }
}

// ------------------------------------------------------- elementwise ops
__global__ void add_resid_kernel(float* __restrict__ x, const float* __restrict__ src,
                                 int rows_x, int rows_s, int d, int total) {
  int i = blockIdx.x * 256 + threadIdx.x;
  if (i >= total) return;
  int c = i % d;
  int r = (i / d) % rows_x;
  int b = i / (d * rows_x);
  x[i] += src[((size_t)b * rows_s + r) * d + c];
}

__global__ void build_xf_kernel(const float* __restrict__ x, const float* __restrict__ regs,
                                float* __restrict__ xf) {
  int i = blockIdx.x * 256 + threadIdx.x;
  if (i >= Bsz * NMEM * Dm) return;
  int c = i % Dm;
  int t = (i / Dm) % NMEM;
  int b = i / (Dm * NMEM);
  float v;
  if (t < NTOK) v = x[((size_t)b * NTOK + t) * Dm + c];
  else if (t < NVAL) v = regs[(t - NTOK) * Dm + c];
  else v = 0.f;
  xf[i] = v;
}

__global__ void lr_act_kernel(float* __restrict__ lr, const float* __restrict__ blr) {
  int i = blockIdx.x * 256 + threadIdx.x;
  if (i >= Bsz * NMEM * Hh) return;
  int hcol = i % Hh;
  int t = (i / Hh) % NMEM;
  float v = sigmoid_f(lr[i] + blr[hcol]);
  lr[i] = (t < NVAL) ? v * 0.01f : 0.f;
}

__global__ void sig_act_kernel(float* __restrict__ a, const float* __restrict__ bias,
                               int total, int cols) {
  int i = blockIdx.x * 256 + threadIdx.x;
  if (i >= total) return;
  a[i] = sigmoid_f(a[i] + bias[i % cols]);
}

__global__ void xm_kernel(const float* __restrict__ xnm, float* __restrict__ xm) {
  int c = blockIdx.x, b = blockIdx.y;
  for (int col = threadIdx.x; col < Dm; col += 256) {
    float s = 0.f;
    for (int t = 0; t < CHm; ++t)
      s += xnm[((size_t)(b * NMEM + c * CHm + t)) * Dm + col];
    xm[((size_t)(b * NCH + c)) * Dm + col] = s * (1.f / 64.f);
  }
}

__global__ void init_mem_kernel(float* __restrict__ M1, float* __restrict__ M2,
                                float* __restrict__ S1, float* __restrict__ S2,
                                const float* __restrict__ M1i, const float* __restrict__ M2i,
                                int total) {
  int i = blockIdx.x * 256 + threadIdx.x;
  if (i >= total) return;
  int j = i & 4095;
  M1[i] = M1i[j];
  M2[i] = M2i[j];
  S1[i] = 0.f;
  S2[i] = 0.f;
}

// ------------------------------- block-level 64x64x64 WMMA matmul helper
__device__ inline void mm64(const bf16_t* A, bool tA, const bf16_t* B, bool tB,
                            float* C) {
  __syncthreads();
  int lane = threadIdx.x & 31;
  int wave = threadIdx.x >> 5;
  for (int t = wave; t < 16; t += 8) {
    int tm = t >> 2, tn = t & 3;
    v8f acc = {};
#pragma unroll
    for (int kk = 0; kk < 64; kk += 32) {
      v16bf fa = frag_a(A, 64, tA, tm * 16, kk, lane);
      v16bf fb = frag_b(B, 64, tB, kk, tn * 16, lane);
      acc = wmma_bf16(fa, fb, acc);
    }
    int n = tn * 16 + (lane & 15);
    int mb = tm * 16 + ((lane >> 4) << 3);
#pragma unroll
    for (int r = 0; r < 8; ++r) C[(mb + r) * 64 + n] = acc[r];
  }
  __syncthreads();
}

// clip_fro + spectral_normalize (3 iters) + negate, all in-place on G (4096 f32)
__device__ inline void process_grad(float* G, float* u, float* vv, float* w, float* red) {
  float ps = 0.f;
  for (int i = threadIdx.x; i < 4096; i += 256) {
    float g = G[i];
    ps += g * g;
  }
  float n2 = block_sum(ps, red);
  float scale = fminf(1.f, 1.0f / (sqrtf(n2) + 1e-8f));  // MAX_GN = 1.0
  for (int i = threadIdx.x; i < 4096; i += 256) G[i] *= scale;
  if (threadIdx.x < 64) vv[threadIdx.x] = 0.125f;  // 1/sqrt(64)
  __syncthreads();
  for (int it = 0; it < 3; ++it) {
    if (threadIdx.x < 64) {
      float s = 0.f;
      for (int e = 0; e < 64; ++e) s += G[threadIdx.x * 64 + e] * vv[e];
      u[threadIdx.x] = s;
    }
    __syncthreads();
    if (threadIdx.x == 0) {
      float s = 0.f;
      for (int i = 0; i < 64; ++i) s += u[i] * u[i];
      red[0] = 1.f / (sqrtf(s) + 1e-8f);
    }
    __syncthreads();
    if (threadIdx.x < 64) u[threadIdx.x] *= red[0];
    __syncthreads();
    if (threadIdx.x < 64) {
      float s = 0.f;
      for (int d = 0; d < 64; ++d) s += G[d * 64 + threadIdx.x] * u[d];
      w[threadIdx.x] = s;
    }
    __syncthreads();
    if (threadIdx.x == 0) {
      float s = 0.f;
      for (int i = 0; i < 64; ++i) s += w[i] * w[i];
      red[0] = 1.f / (sqrtf(s) + 1e-8f);
    }
    __syncthreads();
    if (threadIdx.x < 64) vv[threadIdx.x] = w[threadIdx.x] * red[0];
    __syncthreads();
  }
  if (threadIdx.x < 64) {
    float s = 0.f;
    for (int e = 0; e < 64; ++e) s += G[threadIdx.x * 64 + e] * vv[e];
    w[threadIdx.x] = u[threadIdx.x] * s;
  }
  __syncthreads();
  if (threadIdx.x == 0) {
    float s = 0.f;
    for (int i = 0; i < 64; ++i) s += w[i];
    red[0] = -1.f / (s + 1e-8f);  // fold the negation
  }
  __syncthreads();
  float inv = red[0];
  for (int i = threadIdx.x; i < 4096; i += 256) G[i] *= inv;
  __syncthreads();
}

// ---------------------------------------------- neural-memory scan step
// one block per (head=blockIdx.x, batch=blockIdx.y); 256 threads; dynamic LDS
__global__ void mem_step_kernel(const float* __restrict__ Qg, const float* __restrict__ Kg,
                                const float* __restrict__ Vg, const float* __restrict__ lrg,
                                const float* __restrict__ momg, const float* __restrict__ decg,
                                float* __restrict__ M1, float* __restrict__ M2,
                                float* __restrict__ S1, float* __restrict__ S2,
                                float* __restrict__ outg, int chunk) {
  int h = blockIdx.x, b = blockIdx.y;
  int row0 = b * NMEM + chunk * CHm;
  size_t mb = ((size_t)(b * Hh + h)) * 4096;
  extern __shared__ char smem[];
  bf16_t* Kb = (bf16_t*)smem;
  bf16_t* Vb = Kb + 4096;
  bf16_t* Qb = Vb + 4096;      // first used as d_h1 scratch, later as q
  bf16_t* M1b = Qb + 4096;
  bf16_t* M2b = M1b + 4096;
  bf16_t* A1b = M2b + 4096;
  bf16_t* DPb = A1b + 4096;
  bf16_t* H1b = DPb + 4096;
  float* Cf = (float*)(H1b + 4096);
  float* sm = Cf + 4096;
  float* lrv = sm;         // 64
  float* uu = sm + 64;     // 64
  float* vv2 = sm + 128;   // 64
  float* ww = sm + 192;    // 64
  float* red = sm + 256;   // 64

  // loads
  for (int i = threadIdx.x; i < 4096; i += 256) {
    int t = i >> 6, d = i & 63;
    size_t r = (size_t)(row0 + t) * Im + h * DH + d;
    Kb[i] = (bf16_t)Kg[r];
    Vb[i] = (bf16_t)Vg[r];
    M1b[i] = (bf16_t)M1[mb + i];
    M2b[i] = (bf16_t)M2[mb + i];
  }
  if (threadIdx.x < CHm)
    lrv[threadIdx.x] = lrg[(size_t)(row0 + threadIdx.x) * Hh + h];

  // h1 = kc @ M1 ; a1 = gelu(h1)
  mm64(Kb, false, M1b, false, Cf);
  for (int i = threadIdx.x; i < 4096; i += 256) {
    float v = Cf[i];
    H1b[i] = (bf16_t)v;
    A1b[i] = (bf16_t)gelu_f(v);
  }
  // pred = a1 @ M2 ; dpred = lr * (pred - vc)
  mm64(A1b, false, M2b, false, Cf);
  for (int i = threadIdx.x; i < 4096; i += 256) {
    int t = i >> 6;
    DPb[i] = (bf16_t)(lrv[t] * (Cf[i] - (float)Vb[i]));
  }
  // d_a1 = dpred @ M2^T ; d_h1 = d_a1 * gelu'(h1)  (stashed in Qb)
  mm64(DPb, false, M2b, true, Cf);
  for (int i = threadIdx.x; i < 4096; i += 256)
    Qb[i] = (bf16_t)(Cf[i] * gelu_grad_f((float)H1b[i]));
  // g2 = a1^T @ dpred -> s2 ; update S2 / M2 (global + LDS)
  mm64(A1b, true, DPb, false, Cf);
  process_grad(Cf, uu, vv2, ww, red);
  {
    float mv = momg[(b * NCH + chunk) * Hh + h];
    float dv = decg[(b * NCH + chunk) * Hh + h];
    for (int i = threadIdx.x; i < 4096; i += 256) {
      float s = mv * S2[mb + i] + Cf[i];
      S2[mb + i] = s;
      float m = (1.f - dv) * M2[mb + i] + s;
      M2[mb + i] = m;
      M2b[i] = (bf16_t)m;
    }
  }
  // g1 = kc^T @ d_h1 -> s1 ; update S1 / M1
  mm64(Kb, true, Qb, false, Cf);
  process_grad(Cf, uu, vv2, ww, red);
  {
    float mv = momg[(b * NCH + chunk) * Hh + h];
    float dv = decg[(b * NCH + chunk) * Hh + h];
    for (int i = threadIdx.x; i < 4096; i += 256) {
      float s = mv * S1[mb + i] + Cf[i];
      S1[mb + i] = s;
      float m = (1.f - dv) * M1[mb + i] + s;
      M1[mb + i] = m;
      M1b[i] = (bf16_t)m;
    }
  }
  __syncthreads();
  // out = gelu(qc @ M1_new) @ M2_new
  for (int i = threadIdx.x; i < 4096; i += 256) {
    int t = i >> 6, d = i & 63;
    Qb[i] = (bf16_t)Qg[(size_t)(row0 + t) * Im + h * DH + d];
  }
  mm64(Qb, false, M1b, false, Cf);
  for (int i = threadIdx.x; i < 4096; i += 256)
    A1b[i] = (bf16_t)gelu_f(Cf[i]);
  mm64(A1b, false, M2b, false, Cf);
  for (int i = threadIdx.x; i < 4096; i += 256) {
    int t = i >> 6, d = i & 63;
    outg[(size_t)(row0 + t) * Im + h * DH + d] = Cf[i];
  }
}

// ---------------------------------------------------------------- host
extern "C" void kernel_launch(void* const* d_in, const int* in_sizes, int n_in,
                              void* d_out, int out_size, void* d_ws, size_t ws_size,
                              hipStream_t stream) {
  (void)in_sizes; (void)n_in; (void)out_size; (void)ws_size;
  const float* img   = (const float*)d_in[0];
  const float* pl1s  = (const float*)d_in[1];
  const float* pl1b  = (const float*)d_in[2];
  const float* Wp    = (const float*)d_in[3];
  const float* bp    = (const float*)d_in[4];
  const float* pl2s  = (const float*)d_in[5];
  const float* pl2b  = (const float*)d_in[6];
  const float* cls   = (const float*)d_in[7];
  const float* lnas  = (const float*)d_in[8];
  const float* lnab  = (const float*)d_in[9];
  const float* Wqkv  = (const float*)d_in[10];
  const float* Wao   = (const float*)d_in[11];
  const float* regs  = (const float*)d_in[12];
  const float* lnms  = (const float*)d_in[13];
  const float* lnmb  = (const float*)d_in[14];
  const float* Wmq   = (const float*)d_in[15];
  const float* Wmk   = (const float*)d_in[16];
  const float* Wmv   = (const float*)d_in[17];
  const float* Wmo   = (const float*)d_in[18];
  const float* Wlr   = (const float*)d_in[19];
  const float* blr   = (const float*)d_in[20];
  const float* Wmom  = (const float*)d_in[21];
  const float* bmom  = (const float*)d_in[22];
  const float* Wdec  = (const float*)d_in[23];
  const float* bdec  = (const float*)d_in[24];
  const float* M1i   = (const float*)d_in[25];
  const float* M2i   = (const float*)d_in[26];
  const float* fls   = (const float*)d_in[27];
  const float* flb   = (const float*)d_in[28];
  const float* Wh    = (const float*)d_in[29];
  const float* bh    = (const float*)d_in[30];

  float* ws = (float*)d_ws;
  size_t off = 0;
  auto alloc = [&](size_t n) { float* p = ws + off; off += n; return p; };
  float* patches = alloc((size_t)Bsz * 196 * Dm);
  float* petmp   = alloc((size_t)Bsz * 196 * Dm);
  float* x       = alloc((size_t)Bsz * NTOK * Dm);
  float* xn_a    = alloc((size_t)Bsz * NATT * Dm);
  float* qkv     = alloc((size_t)Bsz * NATT * 3 * Im);
  float* attno   = alloc((size_t)Bsz * NATT * Im);
  float* attnp   = alloc((size_t)Bsz * NATT * Dm);
  float* xf      = alloc((size_t)Bsz * NMEM * Dm);
  float* xnm     = alloc((size_t)Bsz * NMEM * Dm);
  float* mq      = alloc((size_t)Bsz * NMEM * Im);
  float* mk      = alloc((size_t)Bsz * NMEM * Im);
  float* mv      = alloc((size_t)Bsz * NMEM * Im);
  float* lrbuf   = alloc((size_t)Bsz * NMEM * Hh);
  float* xm      = alloc((size_t)Bsz * NCH * Dm);
  float* momb    = alloc((size_t)Bsz * NCH * Hh);
  float* decb    = alloc((size_t)Bsz * NCH * Hh);
  float* M1      = alloc((size_t)Bsz * Hh * 4096);
  float* M2      = alloc((size_t)Bsz * Hh * 4096);
  float* S1      = alloc((size_t)Bsz * Hh * 4096);
  float* S2      = alloc((size_t)Bsz * Hh * 4096);
  float* mraw    = alloc((size_t)Bsz * NMEM * Im);
  float* mproj   = alloc((size_t)Bsz * NMEM * Dm);
  float* clsn    = alloc((size_t)Bsz * Dm);

  const int M_PAT = Bsz * 196;       // 6272
  const int M_ATT = Bsz * NATT;      // 6720
  const int M_MEM = Bsz * NMEM;      // 8192
  const int TOT_X = Bsz * NTOK * Dm; // 4841472
  const size_t STEP_SMEM = 8 * 4096 * sizeof(bf16_t) + 4096 * sizeof(float) + 512 * sizeof(float);

  // ---- patch embedding
  patchify_ln_kernel<<<dim3(196, Bsz), 256, 0, stream>>>(img, pl1s, pl1b, patches);
  gemm_wmma<true><<<dim3((Dm + 63) / 64, (M_PAT + 127) / 128), 256, 0, stream>>>(patches, Wp, bp, petmp, M_PAT, Dm, Dm);
  postpatch_kernel<<<dim3(196, Bsz), 256, 0, stream>>>(petmp, pl2s, pl2b, x);
  write_cls_kernel<<<Bsz, 256, 0, stream>>>(cls, x);

  for (int l = 0; l < Ltot; ++l) {
    // ---- window attention
    layernorm_kernel<<<dim3(NATT, Bsz), 256, 0, stream>>>(x, lnas + l * Dm, lnab + l * Dm, xn_a, NTOK, NATT, NTOK, Dm);
    gemm_wmma<false><<<dim3((3 * Im + 63) / 64, (M_ATT + 127) / 128), 256, 0, stream>>>(
        xn_a, Wqkv + (size_t)l * Dm * 3 * Im, nullptr, qkv, M_ATT, 3 * Im, Dm);
    window_attn_kernel<<<dim3(NWIN, Hh, Bsz), 256, 0, stream>>>(qkv, attno);
    gemm_wmma<false><<<dim3((Dm + 63) / 64, (M_ATT + 127) / 128), 256, 0, stream>>>(
        attno, Wao + (size_t)l * Im * Dm, nullptr, attnp, M_ATT, Dm, Im);
    add_resid_kernel<<<(TOT_X + 255) / 256, 256, 0, stream>>>(x, attnp, NTOK, NATT, Dm, TOT_X);

    // ---- neural memory
    build_xf_kernel<<<(Bsz * NMEM * Dm + 255) / 256, 256, 0, stream>>>(x, regs + (size_t)l * 4 * Dm, xf);
    layernorm_kernel<<<dim3(NMEM, Bsz), 256, 0, stream>>>(xf, lnms + l * Dm, lnmb + l * Dm, xnm, NMEM, NMEM, NVAL, Dm);
    gemm_wmma<false><<<dim3((Im + 63) / 64, M_MEM / 128), 256, 0, stream>>>(xnm, Wmq + (size_t)l * Dm * Im, nullptr, mq, M_MEM, Im, Dm);
    gemm_wmma<false><<<dim3((Im + 63) / 64, M_MEM / 128), 256, 0, stream>>>(xnm, Wmk + (size_t)l * Dm * Im, nullptr, mk, M_MEM, Im, Dm);
    gemm_wmma<false><<<dim3((Im + 63) / 64, M_MEM / 128), 256, 0, stream>>>(xnm, Wmv + (size_t)l * Dm * Im, nullptr, mv, M_MEM, Im, Dm);
    gemm_wmma<false><<<dim3(1, M_MEM / 128), 256, 0, stream>>>(xnm, Wlr + (size_t)l * Dm * Hh, nullptr, lrbuf, M_MEM, Hh, Dm);
    lr_act_kernel<<<(Bsz * NMEM * Hh + 255) / 256, 256, 0, stream>>>(lrbuf, blr + l * Hh);
    xm_kernel<<<dim3(NCH, Bsz), 256, 0, stream>>>(xnm, xm);
    gemm_wmma<false><<<dim3(1, 1), 256, 0, stream>>>(xm, Wmom + (size_t)l * Dm * Hh, nullptr, momb, Bsz * NCH, Hh, Dm);
    gemm_wmma<false><<<dim3(1, 1), 256, 0, stream>>>(xm, Wdec + (size_t)l * Dm * Hh, nullptr, decb, Bsz * NCH, Hh, Dm);
    sig_act_kernel<<<(Bsz * NCH * Hh + 255) / 256, 256, 0, stream>>>(momb, bmom + l * Hh, Bsz * NCH * Hh, Hh);
    sig_act_kernel<<<(Bsz * NCH * Hh + 255) / 256, 256, 0, stream>>>(decb, bdec + l * Hh, Bsz * NCH * Hh, Hh);
    init_mem_kernel<<<(Bsz * Hh * 4096 + 255) / 256, 256, 0, stream>>>(
        M1, M2, S1, S2, M1i + (size_t)l * 4096, M2i + (size_t)l * 4096, Bsz * Hh * 4096);
    for (int c = 0; c < NCH; ++c)
      mem_step_kernel<<<dim3(Hh, Bsz), 256, STEP_SMEM, stream>>>(
          mq, mk, mv, lrbuf, momb, decb, M1, M2, S1, S2, mraw, c);
    gemm_wmma<false><<<dim3((Dm + 63) / 64, M_MEM / 128), 256, 0, stream>>>(
        mraw, Wmo + (size_t)l * Im * Dm, nullptr, mproj, M_MEM, Dm, Im);
    add_resid_kernel<<<(TOT_X + 255) / 256, 256, 0, stream>>>(x, mproj, NTOK, NMEM, Dm, TOT_X);
  }

  // ---- head
  layernorm_kernel<<<dim3(1, Bsz), 256, 0, stream>>>(x, fls, flb, clsn, NTOK, 1, 1, Dm);
  gemm_wmma<true><<<dim3((1000 + 63) / 64, 1), 256, 0, stream>>>(clsn, Wh, bh, (float*)d_out, Bsz, 1000, Dm);
}
